// UpdateEdge_85744727097817
// MI455X (gfx1250) — matrix operand order repaired and checked by reading, not compiled, not run on hardware
//
#include <hip/hip_runtime.h>
#include <hip/hip_bf16.h>
#include <math.h>

#define NSC 64
#define NVC 32
#define DIMF 160
#define LATD 128
#define HIDD 256
#define NTD 4
#define N_NODES 50000
#define N_EDGES 300000
#define N_ACT 200000
#define SQ3F 1.7320508075688772f

typedef __attribute__((ext_vector_type(16))) _Float16 v16h;
typedef __attribute__((ext_vector_type(8)))  _Float16 v8h;
typedef __attribute__((ext_vector_type(8)))  float    v8f;
typedef _Float16 half_t;

// ---------------------------------------------------------------------------
// WMMA fragment helpers (wave32, 16x16x32 f16, f32 accumulate)
// A layout (ISA 7.12.2): lanes 0-15 -> M=0..15, K {0..7}u{16..23};
//                        lanes 16-31 -> same M, K {8..15}u{24..31}.
// B fragments are PRE-PACKED in memory as [ntile][ktile][lane][16] so each
// lane issues one contiguous 32B load (2x global_load_b128, coalesced).
// C/D layout: VGPR r -> M = r + 8*(lane>=16), N = lane&15.
// ---------------------------------------------------------------------------
static __device__ __forceinline__ v16h load_a(const half_t* base, int stride,
                                              int k0, int lane) {
  const int m  = lane & 15;
  const int kb = (lane >> 4) ? 8 : 0;
  const half_t* row = base + m * stride + k0;
  v16h a;
#pragma unroll
  for (int vg = 0; vg < 8; ++vg) {
    int k = (vg < 4) ? (kb + 2 * vg) : (16 + kb + 2 * (vg - 4));
    a[2 * vg]     = row[k];
    a[2 * vg + 1] = row[k + 1];
  }
  return a;
}

static __device__ __forceinline__ v16h load_b_pk(const half_t* __restrict__ frag,
                                                 int ktiles, int ntile, int ktile,
                                                 int lane) {
  return *(const v16h*)(frag + (((size_t)(ntile * ktiles + ktile) * 32 + lane) << 4));
}

static __device__ __forceinline__ void acc_store_lds(half_t* base, int stride,
                                                     int n0, v8f c, int lane) {
  const int nn = n0 + (lane & 15);
  const int mb = (lane >> 4) << 3;
#pragma unroll
  for (int r = 0; r < 8; ++r) base[(mb + r) * stride + nn] = (half_t)c[r];
}

#define WMMA(a, b, c) \
  __builtin_amdgcn_wmma_f32_16x16x32_f16(false, (a), false, (b), (short)0, (c), false, false)

static __device__ __forceinline__ float sigmoidf_(float x) {
  return 1.f / (1.f + __expf(-x));
}

// ---------------------------------------------------------------------------
// Pack a row-major f32 weight (K x N) into f16 WMMA B fragments.
// dst layout: [ntile][ktile][lane][16]; zero-fill for k >= K (ragged K=232).
// ---------------------------------------------------------------------------
__global__ void pack_b(const float* __restrict__ src, half_t* __restrict__ dst,
                       int K, int N) {
  int ktiles = (K + 31) >> 5;
  int total  = ktiles * (N >> 4) * 512;
  int i = blockIdx.x * blockDim.x + threadIdx.x;
  if (i >= total) return;
  int j     = i & 15;
  int lane  = (i >> 4) & 31;
  int ft    = i >> 9;
  int ktile = ft % ktiles;
  int ntile = ft / ktiles;
  int k = ktile * 32 + ((lane >> 4) ? 8 : 0) + (j & 7) + ((j >> 3) << 4);
  int n = ntile * 16 + (lane & 15);
  dst[i] = (k < K) ? (half_t)src[(size_t)k * N + n] : (half_t)0.f;
}

// ---------------------------------------------------------------------------
// latents_out = latents (full copy; active rows overwritten by edge_update)
// ---------------------------------------------------------------------------
__global__ void copy_f4(const float4* __restrict__ src, float4* __restrict__ dst, int n4) {
  int i = blockIdx.x * blockDim.x + threadIdx.x;
  if (i < n4) dst[i] = src[i];
}

// ---------------------------------------------------------------------------
// Scalar LayerNorm (64 ch) + vector RMS norm (32 x 3) per row -> f16 table.
// One wave32 per row; shuffle reductions.
// ---------------------------------------------------------------------------
__global__ __launch_bounds__(256) void sln_rows(const float* __restrict__ x,
                                                const float* __restrict__ gs,
                                                const float* __restrict__ bs,
                                                const float* __restrict__ gv,
                                                half_t* __restrict__ out, int rows) {
  int lane = threadIdx.x & 31;
  int row  = blockIdx.x * 8 + (threadIdx.x >> 5);
  if (row >= rows) return;
  const float* xr = x + (size_t)row * DIMF;
  half_t* orow    = out + (size_t)row * DIMF;

  float a = xr[lane], b = xr[lane + 32];
  float s = a + b, s2 = a * a + b * b;
#pragma unroll
  for (int off = 16; off > 0; off >>= 1) {
    s  += __shfl_xor(s, off, 32);
    s2 += __shfl_xor(s2, off, 32);
  }
  float mu  = s * (1.f / 64.f);
  float inv = rsqrtf(s2 * (1.f / 64.f) - mu * mu + 1e-5f);
  orow[lane]      = (half_t)((a - mu) * inv * gs[lane] + bs[lane]);
  orow[lane + 32] = (half_t)((b - mu) * inv * gs[lane + 32] + bs[lane + 32]);

  float vx = xr[64 + lane * 3 + 0];
  float vy = xr[64 + lane * 3 + 1];
  float vz = xr[64 + lane * 3 + 2];
  float m  = vx * vx + vy * vy + vz * vz;
#pragma unroll
  for (int off = 16; off > 0; off >>= 1) m += __shfl_xor(m, off, 32);
  float g = gv[lane] / sqrtf(m * (1.f / 32.f) + 1e-5f);
  orow[64 + lane * 3 + 0] = (half_t)(vx * g);
  orow[64 + lane * 3 + 1] = (half_t)(vy * g);
  orow[64 + lane * 3 + 2] = (half_t)(vz * g);
}

// ---------------------------------------------------------------------------
// Shared-memory layout (halves). Regions aliased by kernel phase:
//   [0 , 4608)  : Sall[16][192] + Vdot[16][96]   (phases 1-2)
//                 HidH[16][256]                  (phase 7, 4096 <= 4608)
//   [4608, 9216): Vcmp[3][16][96]                (phases 1-2)
//                 RsH[16][64] + RvH[3][16][32]   (phases 4-5, 2560 <= 4608)
//   [9216, ...) : LatH, Latin, MsH, MvH (persistent)
// ---------------------------------------------------------------------------
#define OFF_SALL  0
#define OFF_VDOT  3072
#define OFF_HIDH  0
#define OFF_VCMP  4608
#define OFF_RSH   4608
#define OFF_RVH   5632
#define OFF_LATH  9216
#define OFF_LATIN 11264
#define OFF_MSH   15360
#define OFF_MVH   16384
#define SMEM_HALVES 17920  // 35,840 bytes

__global__ __launch_bounds__(32) void edge_update(
    const float* __restrict__ latents, const float* __restrict__ onehot,
    const float* __restrict__ edge_feat, const float* __restrict__ edge_sh,
    const int* __restrict__ edge_index, const float* __restrict__ cutoff,
    const int* __restrict__ active, const float* __restrict__ ln_g,
    const float* __restrict__ ln_b, const half_t* __restrict__ nfH,
    const half_t* __restrict__ efH, const half_t* __restrict__ Wa,
    const half_t* __restrict__ Wb, const half_t* __restrict__ Wc,
    const half_t* __restrict__ Wd, const half_t* __restrict__ Wps,
    const half_t* __restrict__ Wpv, const half_t* __restrict__ We,
    const half_t* __restrict__ W1, const half_t* __restrict__ W2,
    const half_t* __restrict__ Wrs, const half_t* __restrict__ Wrv,
    float* __restrict__ out_edge, float* __restrict__ out_lat) {
  __shared__ __align__(32) half_t smem[SMEM_HALVES];
  __shared__ float shv[16][4];
  __shared__ float cutS[16];
  __shared__ int   ii[16], jj[16], aa[16];

  half_t (*Sall)[192]     = reinterpret_cast<half_t(*)[192]>(smem + OFF_SALL);
  half_t (*Vdot)[96]      = reinterpret_cast<half_t(*)[96]>(smem + OFF_VDOT);
  half_t (*HidH)[HIDD]    = reinterpret_cast<half_t(*)[HIDD]>(smem + OFF_HIDH);
  half_t (*Vcmp)[16][96]  = reinterpret_cast<half_t(*)[16][96]>(smem + OFF_VCMP);
  half_t (*RsH)[NSC]      = reinterpret_cast<half_t(*)[NSC]>(smem + OFF_RSH);
  half_t (*RvH)[16][NVC]  = reinterpret_cast<half_t(*)[16][NVC]>(smem + OFF_RVH);
  half_t (*LatH)[LATD]    = reinterpret_cast<half_t(*)[LATD]>(smem + OFF_LATH);
  half_t (*Latin)[256]    = reinterpret_cast<half_t(*)[256]>(smem + OFF_LATIN);
  half_t (*MsH)[NSC]      = reinterpret_cast<half_t(*)[NSC]>(smem + OFF_MSH);
  half_t (*MvH)[16][NVC]  = reinterpret_cast<half_t(*)[16][NVC]>(smem + OFF_MVH);

  const int lane = threadIdx.x;
  const int e0   = blockIdx.x * 16;
  if (e0 >= N_ACT) return;

  if (lane < 16) {
    int a      = active[e0 + lane];
    aa[lane]   = a;
    ii[lane]   = edge_index[a];
    jj[lane]   = edge_index[N_EDGES + a];
    cutS[lane] = cutoff[a];
#pragma unroll
    for (int c = 0; c < 4; ++c) shv[lane][c] = edge_sh[(size_t)(e0 + lane) * 4 + c];
  }
  __syncthreads();

  // ---- phase 1: stage gathered features (16B vector loads) ---------------
  for (int t = lane; t < 16 * 20; t += 32) {  // 20 chunks of 8 halves per row
    int e = t / 20, q = t - e * 20;
    int d0 = q * 8;
    v8h vi = *(const v8h*)(nfH + (size_t)ii[e] * DIMF + d0);
    v8h ve = *(const v8h*)(efH + (size_t)(e0 + e) * DIMF + d0);
    v8h vj = *(const v8h*)(nfH + (size_t)jj[e] * DIMF + d0);
    float sh0 = shv[e][0];
    if (q < 8) {
#pragma unroll
      for (int u = 0; u < 8; ++u) {
        int d = d0 + u;
        Sall[e][d]           = (half_t)((float)vi[u] * sh0);
        Sall[e][NSC + d]     = (half_t)((float)ve[u] * sh0);
        Sall[e][2 * NSC + d] = (half_t)((float)vj[u] * sh0);
      }
    } else {
#pragma unroll
      for (int u = 0; u < 8; ++u) {
        int dv = d0 + u - NSC, v = dv / 3, c = dv - v * 3;
        Vcmp[c][e][v]           = vi[u];
        Vcmp[c][e][NVC + v]     = ve[u];
        Vcmp[c][e][2 * NVC + v] = vj[u];
      }
    }
  }
  for (int t = lane; t < 16 * 32; t += 32) {  // latents rows as float4
    int e = t >> 5, q = t & 31;
    float4 f = *(const float4*)(latents + (size_t)aa[e] * LATD + q * 4);
    LatH[e][q * 4 + 0] = (half_t)f.x;
    LatH[e][q * 4 + 1] = (half_t)f.y;
    LatH[e][q * 4 + 2] = (half_t)f.z;
    LatH[e][q * 4 + 3] = (half_t)f.w;
  }
  __syncthreads();

  // vdot = (v_all . sh1)/sqrt(3)
  for (int t = lane; t < 16 * 96; t += 32) {
    int e = t / 96, v = t - e * 96;
    float d = ((float)Vcmp[0][e][v] * shv[e][1] + (float)Vcmp[1][e][v] * shv[e][2] +
               (float)Vcmp[2][e][v] * shv[e][3]) * (1.0f / SQ3F);
    Vdot[e][v] = (half_t)d;
  }

  // lat_in: onehots + zero pad (lane<16), layernorm cooperatively
  if (lane < 16) {
    int e = lane;
#pragma unroll
    for (int c = 0; c < 4; ++c) Latin[e][c] = (half_t)onehot[(size_t)ii[e] * NTD + c];
#pragma unroll
    for (int c = 0; c < 4; ++c) Latin[e][228 + c] = (half_t)onehot[(size_t)jj[e] * NTD + c];
    for (int c = 232; c < 256; ++c) Latin[e][c] = (half_t)0.f;
  }
  __syncthreads();
  for (int e = 0; e < 16; ++e) {  // all 32 lanes reduce one row (4 elems each)
    float x[4], s = 0.f, s2 = 0.f;
#pragma unroll
    for (int q = 0; q < 4; ++q) {
      x[q] = (float)LatH[e][lane * 4 + q];
      s += x[q]; s2 += x[q] * x[q];
    }
#pragma unroll
    for (int off = 16; off > 0; off >>= 1) {
      s  += __shfl_xor(s, off, 32);
      s2 += __shfl_xor(s2, off, 32);
    }
    float mu  = s * (1.f / LATD);
    float inv = rsqrtf(s2 * (1.f / LATD) - mu * mu + 1e-5f);
#pragma unroll
    for (int q = 0; q < 4; ++q)
      Latin[e][4 + lane * 4 + q] =
          (half_t)((x[q] - mu) * inv * ln_g[lane * 4 + q] + ln_b[lane * 4 + q]);
  }
  __syncthreads();

  const int nn = lane & 15;
  const int mb = (lane >> 4) << 3;

  // ---- phase 2: K-outer GEMMs, A fragments hoisted -----------------------
  // o_s = (s_all*sh0)@Wa + vdot@Wb; sWc = (s_all*sh0)@Wc  (share the Sall A stream)
  v8f os[6] = {};
  v8f swc[2] = {};
#pragma unroll
  for (int kt = 0; kt < 6; ++kt) {
    v16h a = load_a(&Sall[0][0], 192, kt * 32, lane);
#pragma unroll
    for (int t = 0; t < 6; ++t) os[t] = WMMA(a, load_b_pk(Wa, 6, t, kt, lane), os[t]);
#pragma unroll
    for (int t = 0; t < 2; ++t) swc[t] = WMMA(a, load_b_pk(Wc, 6, t, kt, lane), swc[t]);
  }
#pragma unroll
  for (int kt = 0; kt < 3; ++kt) {
    v16h a = load_a(&Vdot[0][0], 96, kt * 32, lane);
#pragma unroll
    for (int t = 0; t < 6; ++t) os[t] = WMMA(a, load_b_pk(Wb, 3, t, kt, lane), os[t]);
  }
#pragma unroll
  for (int t = 0; t < 6; ++t)
    acc_store_lds(&Latin[0][0], 256, 132 + t * 16, os[t], lane);  // raw o_s -> lat_in

  // vd_c = v_c@Wd (16x32 per component)
  v8f vd[3][2] = {};
#pragma unroll
  for (int cc = 0; cc < 3; ++cc)
#pragma unroll
    for (int kt = 0; kt < 3; ++kt) {
      v16h a = load_a(&Vcmp[cc][0][0], 96, kt * 32, lane);
#pragma unroll
      for (int t = 0; t < 2; ++t) vd[cc][t] = WMMA(a, load_b_pk(Wd, 3, t, kt, lane), vd[cc][t]);
    }

  // ---- phase 3: m_s = silu(o_s[:, :64]); m_v = o_v * sigmoid(o_s[:,64:]) -
#pragma unroll
  for (int t = 0; t < 4; ++t) {
    v8f ms;
#pragma unroll
    for (int r = 0; r < 8; ++r) { float x = os[t][r]; ms[r] = x * sigmoidf_(x); }
    acc_store_lds(&MsH[0][0], NSC, t * 16, ms, lane);
  }
#pragma unroll
  for (int t = 0; t < 2; ++t) {
#pragma unroll
    for (int r = 0; r < 8; ++r) {
      float g = sigmoidf_(os[4 + t][r]);
      int   m = mb + r;
#pragma unroll
      for (int cc = 0; cc < 3; ++cc) {
        float ov = swc[t][r] * shv[m][1 + cc] + vd[cc][t][r] * shv[m][0];
        MvH[cc][m][t * 16 + nn] = (half_t)(ov * g);
      }
    }
  }
  __syncthreads();

  // ---- phase 4: residual staging (reuses Vcmp region, now dead) ----------
  for (int t = lane; t < 16 * 40; t += 32) {  // 40 float4 chunks per row
    int e = t / 40, q = t - e * 40;
    int d0 = q * 4;
    float4 f = *(const float4*)(edge_feat + (size_t)(e0 + e) * DIMF + d0);
    float vals[4] = {f.x, f.y, f.z, f.w};
    if (q < 16) {
#pragma unroll
      for (int u = 0; u < 4; ++u) RsH[e][d0 + u] = (half_t)vals[u];
    } else {
#pragma unroll
      for (int u = 0; u < 4; ++u) {
        int dv = d0 + u - NSC;
        RvH[dv % 3][e][dv / 3] = (half_t)vals[u];
      }
    }
  }
  __syncthreads();

  // ---- phase 5: projections + ww + residual GEMMs (K-outer) --------------
  v8f ms2[4] = {};
#pragma unroll
  for (int kt = 0; kt < 2; ++kt) {
    v16h a = load_a(&MsH[0][0], NSC, kt * 32, lane);
#pragma unroll
    for (int t = 0; t < 4; ++t) ms2[t] = WMMA(a, load_b_pk(Wps, 2, t, kt, lane), ms2[t]);
  }
  v8f mv2[3][2];
#pragma unroll
  for (int cc = 0; cc < 3; ++cc) {
    v16h a = load_a(&MvH[cc][0][0], NVC, 0, lane);
#pragma unroll
    for (int t = 0; t < 2; ++t) {
      v8f z = {};
      mv2[cc][t] = WMMA(a, load_b_pk(Wpv, 1, t, 0, lane), z);
    }
  }
  v8f ww[6] = {};
#pragma unroll
  for (int kt = 0; kt < 4; ++kt) {
    v16h a = load_a(&LatH[0][0], LATD, kt * 32, lane);
#pragma unroll
    for (int t = 0; t < 6; ++t) ww[t] = WMMA(a, load_b_pk(We, 4, t, kt, lane), ww[t]);
  }
  v8f rs2[4] = {};
#pragma unroll
  for (int kt = 0; kt < 2; ++kt) {
    v16h a = load_a(&RsH[0][0], NSC, kt * 32, lane);
#pragma unroll
    for (int t = 0; t < 4; ++t) rs2[t] = WMMA(a, load_b_pk(Wrs, 2, t, kt, lane), rs2[t]);
  }
  v8f rv2[3][2];
#pragma unroll
  for (int cc = 0; cc < 3; ++cc) {
    v16h a = load_a(&RvH[cc][0][0], NVC, 0, lane);
#pragma unroll
    for (int t = 0; t < 2; ++t) {
      v8f z = {};
      rv2[cc][t] = WMMA(a, load_b_pk(Wrv, 1, t, 0, lane), z);
    }
  }

  // ---- phase 6: edge_out --------------------------------------------------
#pragma unroll
  for (int t = 0; t < 4; ++t)
#pragma unroll
    for (int r = 0; r < 8; ++r) {
      int m = mb + r;
      out_edge[(size_t)(e0 + m) * DIMF + t * 16 + nn] = ms2[t][r] * ww[t][r] + rs2[t][r];
    }
#pragma unroll
  for (int t = 0; t < 2; ++t)
#pragma unroll
    for (int cc = 0; cc < 3; ++cc)
#pragma unroll
      for (int r = 0; r < 8; ++r) {
        int m = mb + r;
        int w = t * 16 + nn;
        out_edge[(size_t)(e0 + m) * DIMF + NSC + w * 3 + cc] =
            mv2[cc][t][r] * ww[4 + t][r] + rv2[cc][t][r];
      }
  __syncthreads();

  // ---- phase 7: latent MLP (HidH aliases Sall/Vdot region), K-outer ------
  {
    v8f h[16] = {};
#pragma unroll
    for (int kt = 0; kt < 8; ++kt) {
      v16h a = load_a(&Latin[0][0], 256, kt * 32, lane);
#pragma unroll
      for (int t = 0; t < 16; ++t) h[t] = WMMA(a, load_b_pk(W1, 8, t, kt, lane), h[t]);
    }
#pragma unroll
    for (int t = 0; t < 16; ++t) {
#pragma unroll
      for (int r = 0; r < 8; ++r) { float x = h[t][r]; h[t][r] = x * sigmoidf_(x); }
      acc_store_lds(&HidH[0][0], HIDD, t * 16, h[t], lane);
    }
  }
  __syncthreads();
  {
    v8f o[8] = {};
#pragma unroll
    for (int kt = 0; kt < 8; ++kt) {
      v16h a = load_a(&HidH[0][0], HIDD, kt * 32, lane);
#pragma unroll
      for (int t = 0; t < 8; ++t) o[t] = WMMA(a, load_b_pk(W2, 8, t, kt, lane), o[t]);
    }
#pragma unroll
    for (int t = 0; t < 8; ++t)
#pragma unroll
      for (int r = 0; r < 8; ++r) {
        int m = mb + r;
        out_lat[(size_t)aa[m] * LATD + t * 16 + nn] = o[t][r] * cutS[m];
      }
  }
}

// ---------------------------------------------------------------------------
extern "C" void kernel_launch(void* const* d_in, const int* in_sizes, int n_in,
                              void* d_out, int out_size, void* d_ws, size_t ws_size,
                              hipStream_t stream) {
  (void)in_sizes; (void)n_in; (void)out_size; (void)ws_size;
  const float* latents   = (const float*)d_in[0];
  const float* node_feat = (const float*)d_in[1];
  const float* onehot    = (const float*)d_in[2];
  const float* edge_feat = (const float*)d_in[3];
  const float* edge_sh   = (const float*)d_in[4];
  const int*   edge_idx  = (const int*)d_in[5];
  const float* cutoff    = (const float*)d_in[6];
  const int*   active    = (const int*)d_in[7];
  const float* n_gs = (const float*)d_in[8];
  const float* n_bs = (const float*)d_in[9];
  const float* n_gv = (const float*)d_in[10];
  const float* e_gs = (const float*)d_in[11];
  const float* e_bs = (const float*)d_in[12];
  const float* e_gv = (const float*)d_in[13];
  const float* ln_g = (const float*)d_in[14];
  const float* ln_b = (const float*)d_in[15];

  // workspace layout (f16): feature tables + pre-packed B fragments
  half_t* ws  = (half_t*)d_ws;
  size_t  off = 0;
  half_t* nfH = ws + off; off += (size_t)N_NODES * DIMF;
  half_t* efH = ws + off; off += (size_t)N_ACT * DIMF;
  // weight (K, N) pairs in d_in[16..26] order
  const int wk[11] = {192, 96, 192, 96, 64, 32, 128, 232, 256, 64, 32};
  const int wn[11] = {96, 96, 32, 32, 64, 32, 96, 256, 128, 64, 32};
  half_t* W[11];
  for (int i = 0; i < 11; ++i) {
    int ktiles = (wk[i] + 31) >> 5;
    int sz     = ktiles * (wn[i] >> 4) * 512;
    W[i] = ws + off; off += (size_t)sz;
    pack_b<<<(sz + 255) / 256, 256, 0, stream>>>((const float*)d_in[16 + i], W[i],
                                                 wk[i], wn[i]);
  }

  // normalized node / edge feature tables (f16)
  sln_rows<<<(N_NODES + 7) / 8, 256, 0, stream>>>(node_feat, n_gs, n_bs, n_gv, nfH, N_NODES);
  sln_rows<<<(N_ACT + 7) / 8, 256, 0, stream>>>(edge_feat, e_gs, e_bs, e_gv, efH, N_ACT);

  float* out_edge = (float*)d_out;
  float* out_lat  = out_edge + (size_t)N_ACT * DIMF;

  // latents_out = latents (active rows overwritten by edge_update)
  int n4 = (N_EDGES * LATD) / 4;
  copy_f4<<<(n4 + 255) / 256, 256, 0, stream>>>((const float4*)latents, (float4*)out_lat, n4);

  // fused edge update: one wave per 16-edge tile
  edge_update<<<N_ACT / 16, 32, 0, stream>>>(
      latents, onehot, edge_feat, edge_sh, edge_idx, cutoff, active, ln_g, ln_b,
      nfH, efH, W[0], W[1], W[2], W[3], W[4], W[5], W[6], W[7], W[8], W[9], W[10],
      out_edge, out_lat);
}
